// SuperResolutionBlock_11854109736985
// MI455X (gfx1250) — matrix operand order repaired and checked by reading, not compiled, not run on hardware
//
#include <hip/hip_runtime.h>
#include <math.h>

// ---------------------------------------------------------------------------
// SuperResolutionBlock on MI455X (gfx1250, wave32, WMMA f16->f32).
//
// All convolutions / einsums are executed as GEMMs on v_wmma_f32_16x16x32_f16:
//   - offset convs (Cout=27, K=576)      : plain im2col taps
//   - DCNv2 einsums (Cout=64, K=576)     : deformable bilinear gather taps
//   - residual 1x1 conv (Cout=64, K=64)  : taps=1, pad=0
// One workgroup = 32 pixels x Cout, 8 waves, each wave a 16x16 WMMA tile.
// ---------------------------------------------------------------------------

typedef __attribute__((ext_vector_type(16))) _Float16 v16h;
typedef __attribute__((ext_vector_type(8)))  float    v8f;

#define CIN      64
#define MAXTAPS  9
#define MAXCH    18          // max K-chunks: 64*9/32
#define NPMAX    (MAXTAPS*32)

// ---------------------------------------------------------------------------
// Pack f32 weight matrix (Cout x Ktot, row-major) into f16 A-fragments in the
// CDNA5 16-bit A-matrix (16x32) lane layout, padded with zeros.
// dst element index: ((ot*chunks + q)*32 + lane)*16 + j
//   row = ot*16 + lane%16 ; K = q*32 + (lane>=16 ? 8:0) + (j<8 ? j : j+8)
// ---------------------------------------------------------------------------
__global__ void pack_weights(const float* __restrict__ src, _Float16* __restrict__ dst,
                             int Cout, int Ktot, int octiles, int chunks) {
  int total = octiles * chunks * 512;
  for (int idx = blockIdx.x * blockDim.x + threadIdx.x; idx < total;
       idx += gridDim.x * blockDim.x) {
    int j    = idx & 15;
    int lane = (idx >> 4) & 31;
    int q    = (idx >> 9) % chunks;
    int ot   = idx / (chunks << 9);
    int row  = ot * 16 + (lane & 15);
    int K    = q * 32 + ((lane >> 4) ? 8 : 0) + (j < 8 ? j : j + 8);
    float v  = (row < Cout && K < Ktot) ? src[row * Ktot + K] : 0.0f;
    dst[idx] = (_Float16)v;
  }
}

// ---------------------------------------------------------------------------
// Unified im2col / deformable-gather GEMM.
//   off == nullptr : plain integer taps (regular conv, zero padding)
//   off != nullptr : DCNv2 (dy = off[2k], dx = off[2k+1], mask = sigmoid(off[18+k]))
//   res != nullptr : fuse 2x bilinear-upsampled residual add after ELU.
// ---------------------------------------------------------------------------
__global__ __launch_bounds__(256)
void dcn_gemm(const float* __restrict__ in, int H, int W, int HW,
              const float* __restrict__ off,
              const _Float16* __restrict__ pA,
              const float* __restrict__ bias,
              float* __restrict__ out,
              int Cout, int octiles, int taps, int pad, int do_elu,
              const float* __restrict__ res, int resH, int resW) {
  __shared__ __align__(32) _Float16 sB[MAXCH * 2 * 32 * 16];   // 36 KB
  __shared__ int   sY0[NPMAX], sY1[NPMAX], sX0[NPMAX], sX1[NPMAX], sBofs[NPMAX];
  __shared__ float sW00[NPMAX], sW01[NPMAX], sW10[NPMAX], sW11[NPMAX];

  const int tid      = threadIdx.x;
  const int tileBase = blockIdx.x * 32;        // 32 consecutive pixels (one row)
  const int npairs   = taps * 32;

  // ---- Phase 1: per-(pixel,tap) bilinear coords + folded corner weights ----
  for (int pr = tid; pr < npairs; pr += 256) {
    int k   = pr >> 5;
    int p   = pr & 31;
    int g   = tileBase + p;
    int b   = g / HW;
    int rem = g - b * HW;
    int yy  = rem / W;
    int xx  = rem - yy * W;
    float dy = 0.f, dx = 0.f, msk = 1.f;
    if (off) {
      const float* ob = off + (size_t)b * 27 * HW + rem;
      dy = ob[(2 * k) * HW];
      dx = ob[(2 * k + 1) * HW];
      float mv = ob[(18 + k) * HW];
      msk = 1.0f / (1.0f + expf(-mv));
    }
    int ki = k / 3, kj = k - (k / 3) * 3;
    float sy  = (float)(yy - pad + ki) + dy;
    float sx  = (float)(xx - pad + kj) + dx;
    float y0f = floorf(sy), x0f = floorf(sx);
    float wy  = sy - y0f,   wx  = sx - x0f;
    int y0 = (int)y0f, x0 = (int)x0f;
    int y1 = y0 + 1,   x1 = x0 + 1;
    float vy0 = (y0 >= 0 && y0 < H) ? 1.f : 0.f;
    float vy1 = (y1 >= 0 && y1 < H) ? 1.f : 0.f;
    float vx0 = (x0 >= 0 && x0 < W) ? 1.f : 0.f;
    float vx1 = (x1 >= 0 && x1 < W) ? 1.f : 0.f;
    sW00[pr] = (1.f - wy) * (1.f - wx) * msk * vy0 * vx0;
    sW01[pr] = (1.f - wy) * wx        * msk * vy0 * vx1;
    sW10[pr] = wy        * (1.f - wx) * msk * vy1 * vx0;
    sW11[pr] = wy        * wx         * msk * vy1 * vx1;
    sY0[pr]  = min(max(y0, 0), H - 1) * W;
    sY1[pr]  = min(max(y1, 0), H - 1) * W;
    sX0[pr]  = min(max(x0, 0), W - 1);
    sX1[pr]  = min(max(x1, 0), W - 1);
    sBofs[pr] = b * CIN * HW;
  }
  __syncthreads();

  // ---- Phase 2: gather samples -> f16 B-fragments in LDS -------------------
  // B (32x16) layout: lane = ((K%32/8)&1 ? 16:0) + N ; j = K%8 + (K%32>=16 ? 8:0) ; N = pixel%16
  const int nelem = CIN * npairs;
  for (int e = tid; e < nelem; e += 256) {
    int c  = e / npairs;
    int pr = e - c * npairs;
    int k  = pr >> 5;
    int p  = pr & 31;
    const float* base = in + sBofs[pr] + c * HW;
    float v = sW00[pr] * base[sY0[pr] + sX0[pr]]
            + sW01[pr] * base[sY0[pr] + sX1[pr]]
            + sW10[pr] * base[sY1[pr] + sX0[pr]]
            + sW11[pr] * base[sY1[pr] + sX1[pr]];
    int K    = c * taps + k;
    int q    = K >> 5;
    int r    = K & 31;
    int lane = (((r >> 3) & 1) << 4) | (p & 15);
    int j    = (r & 7) | ((r >> 4) << 3);
    int sub  = p >> 4;
    sB[(((q * 2 + sub) * 32) + lane) * 16 + j] = (_Float16)v;
  }
  __syncthreads();

  // ---- Phase 3: WMMA mainloop + fused epilogue -----------------------------
  const int wave   = tid >> 5;
  const int lane   = tid & 31;
  const int ot     = wave >> 1;                 // oc tile (16 rows)
  const int st     = wave & 1;                  // pixel sub-tile (16 cols)
  const int chunks = (CIN * taps) >> 5;
  if (ot < octiles) {
    v8f acc = {};
    const v16h* Aptr = (const v16h*)pA + (size_t)(ot * chunks) * 32 + lane;
    const v16h* Bptr = (const v16h*)sB;
    for (int q = 0; q < chunks; ++q) {
      v16h a  = Aptr[q * 32];
      v16h bm = Bptr[(q * 2 + st) * 32 + lane];
      acc = __builtin_amdgcn_wmma_f32_16x16x32_f16(false, a, false, bm,
                                                   (short)0, acc, false, false);
    }
    int p   = st * 16 + (lane & 15);
    int g   = tileBase + p;
    int b   = g / HW;
    int rem = g - b * HW;
    int yy  = rem / W;
    int xx  = rem - yy * W;
    int mb  = (lane >> 4) * 8;                  // C/D layout: lanes16-31 -> M+8
#pragma unroll
    for (int i = 0; i < 8; ++i) {
      int oc = ot * 16 + mb + i;
      if (oc < Cout) {
        float v = acc[i] + bias[oc];
        if (do_elu) v = (v > 0.f) ? v : expm1f(v);
        if (res) {   // fused: += bilinear-2x-upsample(res)[b, oc, yy, xx]
          float fy = yy * 0.5f - 0.25f;
          float fx = xx * 0.5f - 0.25f;
          float y0f = floorf(fy), x0f = floorf(fx);
          float wy = fy - y0f,    wx = fx - x0f;
          int ry0 = min(max((int)y0f,     0), resH - 1);
          int ry1 = min(max((int)y0f + 1, 0), resH - 1);
          int rx0 = min(max((int)x0f,     0), resW - 1);
          int rx1 = min(max((int)x0f + 1, 0), resW - 1);
          const float* rp = res + (size_t)(b * Cout + oc) * resH * resW;
          v += (1.f - wy) * ((1.f - wx) * rp[ry0 * resW + rx0] + wx * rp[ry0 * resW + rx1])
             +        wy  * ((1.f - wx) * rp[ry1 * resW + rx0] + wx * rp[ry1 * resW + rx1]);
        }
        out[(size_t)(b * Cout + oc) * HW + rem] = v;
      }
    }
  }
}

// ---------------------------------------------------------------------------
// 2x bilinear upsample (half-pixel centers, edge clamp) — bandwidth kernel.
// ---------------------------------------------------------------------------
__global__ void upsample2x_k(const float* __restrict__ in, float* __restrict__ out,
                             int BC, int H, int W) {
  int OH = 2 * H, OW = 2 * W;
  int total = BC * OH * OW;
  for (int idx = blockIdx.x * blockDim.x + threadIdx.x; idx < total;
       idx += gridDim.x * blockDim.x) {
    int x  = idx % OW;
    int t  = idx / OW;
    int y  = t % OH;
    int bc = t / OH;
    float fy = y * 0.5f - 0.25f;
    float fx = x * 0.5f - 0.25f;
    float y0f = floorf(fy), x0f = floorf(fx);
    float wy = fy - y0f,    wx = fx - x0f;
    int y0 = min(max((int)y0f,     0), H - 1);
    int y1 = min(max((int)y0f + 1, 0), H - 1);
    int x0 = min(max((int)x0f,     0), W - 1);
    int x1 = min(max((int)x0f + 1, 0), W - 1);
    const float* p = in + (size_t)bc * H * W;
    out[idx] = (1.f - wy) * ((1.f - wx) * p[y0 * W + x0] + wx * p[y0 * W + x1])
             +        wy  * ((1.f - wx) * p[y1 * W + x0] + wx * p[y1 * W + x1]);
  }
}

// ---------------------------------------------------------------------------
extern "C" void kernel_launch(void* const* d_in, const int* in_sizes, int n_in,
                              void* d_out, int out_size, void* d_ws, size_t ws_size,
                              hipStream_t stream) {
  // inputs: x, ow1, ob1, w1, b1, ow2, ob2, w2, b2, rw, rb (all f32)
  const float* x   = (const float*)d_in[0];
  const float* ow1 = (const float*)d_in[1];
  const float* ob1 = (const float*)d_in[2];
  const float* w1  = (const float*)d_in[3];
  const float* b1  = (const float*)d_in[4];
  const float* ow2 = (const float*)d_in[5];
  const float* ob2 = (const float*)d_in[6];
  const float* w2  = (const float*)d_in[7];
  const float* b2  = (const float*)d_in[8];
  const float* rw  = (const float*)d_in[9];
  const float* rb  = (const float*)d_in[10];

  const int B = 2, H1 = 192, W1 = 192, H2 = 384, W2 = 384;
  const int HW1 = H1 * W1, HW2 = H2 * W2;
  const int P1 = B * HW1, P2 = B * HW2;

  // workspace carving (256B aligned)
  size_t cur = 0;
  auto carve = [&](size_t bytes) -> void* {
    void* p = (char*)d_ws + cur;
    cur += (bytes + 255) & ~(size_t)255;
    return p;
  };
  _Float16* pOW1 = (_Float16*)carve((size_t)2 * 18 * 512 * 2);
  _Float16* pW1  = (_Float16*)carve((size_t)4 * 18 * 512 * 2);
  _Float16* pOW2 = (_Float16*)carve((size_t)2 * 18 * 512 * 2);
  _Float16* pW2  = (_Float16*)carve((size_t)4 * 18 * 512 * 2);
  _Float16* pRW  = (_Float16*)carve((size_t)4 * 2  * 512 * 2);
  float* off1 = (float*)carve((size_t)B * 27 * HW1 * 4);
  float* h1   = (float*)carve((size_t)B * 64 * HW1 * 4);
  float* up1  = (float*)carve((size_t)B * 64 * HW2 * 4);
  float* off2 = (float*)carve((size_t)B * 27 * HW2 * 4);
  float* res  = (float*)carve((size_t)B * 64 * HW1 * 4);

  // ---- pack all weights into f16 WMMA A-fragments ----
  pack_weights<<<72, 256, 0, stream>>>(ow1, pOW1, 27, 576, 2, 18);
  pack_weights<<<144, 256, 0, stream>>>(w1,  pW1,  64, 576, 4, 18);
  pack_weights<<<72, 256, 0, stream>>>(ow2, pOW2, 27, 576, 2, 18);
  pack_weights<<<144, 256, 0, stream>>>(w2,  pW2,  64, 576, 4, 18);
  pack_weights<<<16, 256, 0, stream>>>(rw,  pRW,  64, 64,  4, 2);

  // ---- residual 1x1 conv at 192^2 (WMMA, taps=1, pad=0) ----
  dcn_gemm<<<P1 / 32, 256, 0, stream>>>(x, H1, W1, HW1, nullptr, pRW, rb, res,
                                        64, 4, 1, 0, 0, nullptr, 0, 0);

  // ---- offset conv 1 (plain 3x3, 27 ch) ----
  dcn_gemm<<<P1 / 32, 256, 0, stream>>>(x, H1, W1, HW1, nullptr, pOW1, ob1, off1,
                                        27, 2, 9, 1, 0, nullptr, 0, 0);

  // ---- DCNv2 #1 + bias + ELU -> h1 ----
  dcn_gemm<<<P1 / 32, 256, 0, stream>>>(x, H1, W1, HW1, off1, pW1, b1, h1,
                                        64, 4, 9, 1, 1, nullptr, 0, 0);

  // ---- 2x bilinear upsample -> up1 ----
  upsample2x_k<<<(B * 64 * HW2 + 255) / 256, 256, 0, stream>>>(h1, up1, B * 64, H1, W1);

  // ---- offset conv 2 (plain 3x3 at 384^2) ----
  dcn_gemm<<<P2 / 32, 256, 0, stream>>>(up1, H2, W2, HW2, nullptr, pOW2, ob2, off2,
                                        27, 2, 9, 1, 0, nullptr, 0, 0);

  // ---- DCNv2 #2 + bias + ELU + fused upsampled-residual add -> d_out ----
  dcn_gemm<<<P2 / 32, 256, 0, stream>>>(up1, H2, W2, HW2, off2, pW2, b2, (float*)d_out,
                                        64, 4, 9, 1, 1, res, H1, W1);
  (void)in_sizes; (void)n_in; (void)out_size; (void)ws_size;
}
